// LSTMBasedTextDecoder_8151847928145
// MI455X (gfx1250) — matrix operand order repaired and checked by reading, not compiled, not run on hardware
//
#include <hip/hip_runtime.h>
#include <hip/hip_bf16.h>
#include <stdint.h>

// ---------------------------------------------------------------------------
// LSTM text decoder for MI455X (gfx1250), bf16 WMMA path + async-to-LDS,
// recurrence distributed over 16 WGPs with a device-scope software barrier.
// B=32, T=128, E=256, H=512, ENC=512, V=10000
// ---------------------------------------------------------------------------

typedef __attribute__((ext_vector_type(16))) __bf16 v16bf;
typedef __attribute__((ext_vector_type(8)))  __bf16 v8bf;
typedef __attribute__((ext_vector_type(8)))  float  v8f;

#define BB   32
#define TT   128
#define EE   256
#define HH   512
#define ENCC 512
#define VV   10000
#define G4   2048   // 4*H
#define NWG  16     // workgroups cooperating on the recurrence

// ---- gfx1250 async copy: global -> LDS, 16 bytes per lane -----------------
__device__ __forceinline__ void async_copy_b128(uint32_t lds_off,
                                                const void* gaddr) {
  asm volatile("global_load_async_to_lds_b128 %0, %1, off"
               :: "v"(lds_off), "v"(gaddr)
               : "memory");
}
__device__ __forceinline__ void wait_async0() {
  asm volatile("s_wait_asynccnt 0x0" ::: "memory");
}

// ---- WMMA tile helpers ----------------------------------------------------
// A (16x32 bf16, row = M): lane l<16 holds row l, K {0..15};
// lane l+16 holds row l, K {16..31}.
__device__ __forceinline__ v16bf load_a_tile(const __bf16* A, int lda,
                                             int m0, int k0, int lane) {
  int row = m0 + (lane & 15);
  int kb  = k0 + ((lane & 16) ? 16 : 0);
  v8bf a0 = *(const v8bf*)(A + (long)row * lda + kb);
  v8bf a1 = *(const v8bf*)(A + (long)row * lda + kb + 8);
  v16bf r;
#pragma unroll
  for (int i = 0; i < 8; ++i) { r[i] = a0[i]; r[i + 8] = a1[i]; }
  return r;
}

// B (32x16 bf16, col = N), B[k][n] = W[n][k] with W row-major [N, K]:
// lane l<16 holds column l, K {0..15}; lane l+16 column l, K {16..31}.
__device__ __forceinline__ v16bf load_bt_tile(const __bf16* W, int ldw,
                                              int n0, int k0, int lane) {
  int col = n0 + (lane & 15);
  int kb  = k0 + ((lane & 16) ? 16 : 0);
  return *(const v16bf*)(W + (long)col * ldw + kb);
}

// ---- conversion / gather kernels ------------------------------------------
__global__ void k_f32_to_bf16(const float* __restrict__ src,
                              __bf16* __restrict__ dst,
                              int src_ld, int cols) {
  long r = blockIdx.x;
  for (int c = threadIdx.x; c < cols; c += blockDim.x)
    dst[r * cols + c] = (__bf16)src[r * src_ld + c];
}

// Y[t*B+b][:] = emb[ t==0 ? start : texts[b][t-1] ][:]
__global__ void k_buildY(const int* __restrict__ texts,
                         const int* __restrict__ start_idx,
                         const float* __restrict__ emb,
                         __bf16* __restrict__ Y) {
  int tb = blockIdx.x;          // t*B + b
  int t  = tb >> 5;
  int b  = tb & 31;
  int tok = (t == 0) ? start_idx[0] : texts[b * TT + (t - 1)];
  int e = threadIdx.x;          // E == blockDim.x == 256
  Y[(long)tb * EE + e] = (__bf16)emb[(long)tok * EE + e];
}

// o = tv@Wo.T+bo ; h0 = tv@Wh.T+bh ; c0 = tv@Wc.T+bc   (f32, tiny)
__global__ __launch_bounds__(512)
void k_init3(const float* __restrict__ tv,
             const float* __restrict__ Wo, const float* __restrict__ bo,
             const float* __restrict__ Wh, const float* __restrict__ bh,
             const float* __restrict__ Wc, const float* __restrict__ bc,
             float* __restrict__ o, float* __restrict__ h, float* __restrict__ c) {
  int which = blockIdx.x / BB;
  int b     = blockIdx.x % BB;
  int j     = threadIdx.x;      // H == 512
  const float* W; const float* bias; float* out;
  if (which == 0)      { W = Wo; bias = bo; out = o; }
  else if (which == 1) { W = Wh; bias = bh; out = h; }
  else                 { W = Wc; bias = bc; out = c; }
  float s = bias[j];
  const float* tvb = tv + (long)b * ENCC;
  const float* wr  = W + (long)j * ENCC;
  for (int k = 0; k < ENCC; ++k) s += tvb[k] * wr[k];
  out[(long)b * HH + j] = s;
}

// const_gates[b][g] = b_ih[g]+b_hh[g] + o[b]·W_ih[g,E:E+H] + tv[b]·W_ih[g,E+H:]
__global__ __launch_bounds__(1024)
void k_constgates(const float* __restrict__ tv, const float* __restrict__ o,
                  const float* __restrict__ W_ih,
                  const float* __restrict__ b_ih, const float* __restrict__ b_hh,
                  float* __restrict__ cg) {
  int b = blockIdx.x >> 1;
  int g = ((blockIdx.x & 1) << 10) + threadIdx.x;     // 0..2047
  const float* wr = W_ih + (long)g * (EE + HH + ENCC);
  float s = b_ih[g] + b_hh[g];
  const float* ob  = o  + (long)b * HH;
  const float* tvb = tv + (long)b * ENCC;
  for (int k = 0; k < HH;   ++k) s += ob[k]  * wr[EE + k];
  for (int k = 0; k < ENCC; ++k) s += tvb[k] * wr[EE + HH + k];
  cg[(long)b * G4 + g] = s;
}

// h0 -> bf16 double-buffer slot 0; reset the software barrier counter
__global__ void k_prep(const float* __restrict__ h0,
                       __bf16* __restrict__ hbuf, unsigned* __restrict__ ctr) {
  int i = blockIdx.x * blockDim.x + threadIdx.x;
  if (i < BB * HH) hbuf[i] = (__bf16)h0[i];
  if (i == 0) *ctr = 0u;
}

// ---- pre_gates = Y[4096,256] @ Wy^T[256,2048]  (bf16 WMMA, f32 out) --------
// Block = 1 M-tile (A async-staged in LDS) x 32 N-tiles; 8 waves x 4 tiles.
__global__ __launch_bounds__(256)
void k_pregates(const __bf16* __restrict__ Y, const __bf16* __restrict__ Wy,
                float* __restrict__ pre) {
  __shared__ __align__(16) __bf16 a_sh[16 * EE];   // 8 KB, contiguous A tile
  int tid  = threadIdx.x;
  int lane = tid & 31;
  int wave = tid >> 5;
  int m0 = blockIdx.y * 16;

  {
    uint32_t lds_base = (uint32_t)(uintptr_t)a_sh;
    const char* gbase = (const char*)(Y + (long)m0 * EE);
#pragma unroll
    for (int j = 0; j < 2; ++j) {
      int boff = (j * 256 + tid) * 16;
      async_copy_b128(lds_base + boff, gbase + boff);
    }
  }
  wait_async0();
  __syncthreads();

  int nt0 = blockIdx.x * 32 + wave * 4;
  v8f acc[4] = {v8f{}, v8f{}, v8f{}, v8f{}};
#pragma unroll
  for (int k0 = 0; k0 < EE; k0 += 32) {
    v16bf a = load_a_tile(a_sh, EE, 0, k0, lane);
#pragma unroll
    for (int c = 0; c < 4; ++c) {
      v16bf b = load_bt_tile(Wy, EE, (nt0 + c) * 16, k0, lane);
      acc[c] = __builtin_amdgcn_wmma_f32_16x16x32_bf16(
          false, a, false, b, (short)0, acc[c], false, false);
    }
  }
#pragma unroll
  for (int c = 0; c < 4; ++c) {
    int n0 = (nt0 + c) * 16;
#pragma unroll
    for (int r = 0; r < 8; ++r) {
      int row = m0 + ((lane < 16) ? r : r + 8);
      int n   = n0 + (lane & 15);
      pre[(long)row * G4 + n] = acc[c][r];
    }
  }
}

// ---- recurrence across NWG workgroups --------------------------------------
// Block g owns hidden columns [g*32, g*32+32) -> 128 gate columns (4 gates).
// 128 threads = 4 waves: wave = (jt<<1)|mt; each wave computes the 4 gate
// tiles of one (m-tile, j-tile), so i/f/g/o of each (b,j) share lane+reg.
// h is double-buffered in global (L2-hot); one release->arrive->spin barrier
// per step; counter is monotonic, reset by k_prep every launch.
__global__ __launch_bounds__(128)
void k_lstm_mw(const float* __restrict__ pre_gates,
               const float* __restrict__ const_gates,
               const __bf16* __restrict__ Whh,
               const float* __restrict__ c0,
               __bf16* __restrict__ hbuf,      // [2][B*H] bf16, slot0 = h0
               unsigned* __restrict__ ctr,
               __bf16* __restrict__ hs) {
  int tid  = threadIdx.x;
  int lane = tid & 31;
  int wave = tid >> 5;
  int g    = blockIdx.x;        // 0..15
  int mt   = wave & 1;
  int jt   = wave >> 1;         // 0..1
  int j0   = g * 32 + jt * 16;

  float c_reg[8];
#pragma unroll
  for (int r = 0; r < 8; ++r) {
    int m = mt * 16 + ((lane < 16) ? r : r + 8);
    c_reg[r] = c0[(long)m * HH + j0 + (lane & 15)];
  }

  for (int t = 0; t < TT; ++t) {
    const __bf16* hcur  = hbuf + (size_t)(t & 1) * (BB * HH);
    __bf16*       hnext = hbuf + (size_t)((t + 1) & 1) * (BB * HH);

    // prefetch first 32 KB (rows 0..3) of next step's pre_gates slice, same for every block
    if (t + 1 < TT) {
      const char* nxt = (const char*)(pre_gates + (long)(t + 1) * BB * G4);
      __builtin_prefetch(nxt + (long)tid * 256, 0, 1);
    }

    v8f acc[4];
#pragma unroll
    for (int ga = 0; ga < 4; ++ga) {
#pragma unroll
      for (int r = 0; r < 8; ++r) {
        int m = mt * 16 + ((lane < 16) ? r : r + 8);
        int n = ga * HH + j0 + (lane & 15);
        acc[ga][r] = pre_gates[((long)t * BB + m) * G4 + n]
                   + const_gates[(long)m * G4 + n];
      }
    }
    // gates += h @ Whh^T  (K = 512), h read from L2-resident double buffer
#pragma unroll 4
    for (int k0 = 0; k0 < HH; k0 += 32) {
      v16bf a = load_a_tile(hcur, HH, mt * 16, k0, lane);
#pragma unroll
      for (int ga = 0; ga < 4; ++ga) {
        v16bf b = load_bt_tile(Whh, HH, ga * HH + j0, k0, lane);
        acc[ga] = __builtin_amdgcn_wmma_f32_16x16x32_bf16(
            false, a, false, b, (short)0, acc[ga], false, false);
      }
    }
    // pointwise LSTM cell on this block's columns
#pragma unroll
    for (int r = 0; r < 8; ++r) {
      int m = mt * 16 + ((lane < 16) ? r : r + 8);
      int j = j0 + (lane & 15);
      float ig = acc[0][r], fg = acc[1][r], gg = acc[2][r], og = acc[3][r];
      float si = 1.f / (1.f + __expf(-ig));
      float sf = 1.f / (1.f + __expf(-fg));
      float so = 1.f / (1.f + __expf(-og));
      float cn = sf * c_reg[r] + si * tanhf(gg);
      c_reg[r] = cn;
      float hn = so * tanhf(cn);
      __bf16 hb = (__bf16)hn;
      hnext[(long)m * HH + j] = hb;
      hs[((long)t * BB + m) * HH + j] = hb;
    }
    // device-scope software barrier (release h, arrive, spin, acquire)
    __threadfence();
    __syncthreads();
    if (tid == 0) {
      atomicAdd(ctr, 1u);
      unsigned target = (unsigned)NWG * (unsigned)(t + 1);
      while (__hip_atomic_load(ctr, __ATOMIC_ACQUIRE,
                               __HIP_MEMORY_SCOPE_AGENT) < target)
        __builtin_amdgcn_s_sleep(1);
    }
    __syncthreads();
    __threadfence();
  }
}

// ---- logits = hs[4096,512] @ Wv^T[512,10000] + bias, scattered to [b][t][v]
__global__ __launch_bounds__(256)
void k_logits(const __bf16* __restrict__ hs, const __bf16* __restrict__ Wv,
              const float* __restrict__ Wv_b, float* __restrict__ out) {
  __shared__ __align__(16) __bf16 a_sh[16 * HH];   // 16 KB
  int tid  = threadIdx.x;
  int lane = tid & 31;
  int wave = tid >> 5;
  int m0 = blockIdx.y * 16;

  {
    uint32_t lds_base = (uint32_t)(uintptr_t)a_sh;
    const char* gbase = (const char*)(hs + (long)m0 * HH);
#pragma unroll
    for (int j = 0; j < 4; ++j) {
      int boff = (j * 256 + tid) * 16;
      async_copy_b128(lds_base + boff, gbase + boff);
    }
  }
  wait_async0();
  __syncthreads();

  int nt0 = blockIdx.x * 32 + wave * 4;
  v8f acc[4] = {v8f{}, v8f{}, v8f{}, v8f{}};
#pragma unroll 4
  for (int k0 = 0; k0 < HH; k0 += 32) {
    v16bf a = load_a_tile(a_sh, HH, 0, k0, lane);
#pragma unroll
    for (int c = 0; c < 4; ++c) {
      if (nt0 + c < 625) {            // wave-uniform guard
        v16bf b = load_bt_tile(Wv, HH, (nt0 + c) * 16, k0, lane);
        acc[c] = __builtin_amdgcn_wmma_f32_16x16x32_bf16(
            false, a, false, b, (short)0, acc[c], false, false);
      }
    }
  }
#pragma unroll
  for (int c = 0; c < 4; ++c) {
    if (nt0 + c < 625) {
      int n0 = (nt0 + c) * 16;
      int v  = n0 + (lane & 15);
      float bias = Wv_b[v];
#pragma unroll
      for (int r = 0; r < 8; ++r) {
        int row = m0 + ((lane < 16) ? r : r + 8);   // row = t*B + b
        int t = row >> 5;
        int b = row & 31;
        out[((long)b * TT + t) * (long)VV + v] = acc[c][r] + bias;
      }
    }
  }
}

// ---------------------------------------------------------------------------
extern "C" void kernel_launch(void* const* d_in, const int* in_sizes, int n_in,
                              void* d_out, int out_size, void* d_ws, size_t ws_size,
                              hipStream_t stream) {
  (void)in_sizes; (void)n_in; (void)out_size; (void)ws_size;
  const float* tv        = (const float*)d_in[0];
  const int*   texts     = (const int*)d_in[1];
  /* d_in[2] = lengths (unused) */
  const int*   start_idx = (const int*)d_in[3];
  const float* emb       = (const float*)d_in[4];
  const float* W_ih      = (const float*)d_in[5];
  const float* b_ih      = (const float*)d_in[6];
  const float* W_hh      = (const float*)d_in[7];
  const float* b_hh      = (const float*)d_in[8];
  const float* Wo_w      = (const float*)d_in[9];
  const float* Wo_b      = (const float*)d_in[10];
  const float* Wh_w      = (const float*)d_in[11];
  const float* Wh_b      = (const float*)d_in[12];
  const float* Wc_w      = (const float*)d_in[13];
  const float* Wc_b      = (const float*)d_in[14];
  const float* Wv_w      = (const float*)d_in[15];
  const float* Wv_b      = (const float*)d_in[16];
  float* out = (float*)d_out;

  char* ws = (char*)d_ws;
  float*    o_buf   = (float*)(ws + 0);          //  32*512*4      = 64 KB
  float*    cg_buf  = (float*)(ws + 65536);      //  32*2048*4     = 256 KB
  float*    c_buf   = (float*)(ws + 327680);     //  32*512*4
  float*    h_buf   = (float*)(ws + 393216);     //  32*512*4
  __bf16*   Y_bf    = (__bf16*)(ws + 458752);    //  4096*256*2    = 2 MB
  __bf16*   Wy_bf   = (__bf16*)(ws + 2555904);   //  2048*256*2    = 1 MB
  __bf16*   Whh_bf  = (__bf16*)(ws + 3604480);   //  2048*512*2    = 2 MB
  __bf16*   Wv_bf   = (__bf16*)(ws + 5701632);   //  10000*512*2   = 10.24 MB
  float*    pre_g   = (float*)(ws + 15941632);   //  4096*2048*4   = 33.5 MB
  __bf16*   hs_bf   = (__bf16*)(ws + 49496064);  //  4096*512*2    = 4 MB
  __bf16*   h_db    = (__bf16*)(ws + 53690368);  //  2*32*512*2    = 64 KB
  unsigned* bar_ctr = (unsigned*)(ws + 53755904);//  4 B
                                                 //  total ~53.8 MB

  // weight conversions to bf16
  k_f32_to_bf16<<<G4, 256, 0, stream>>>(W_ih, Wy_bf, EE + HH + ENCC, EE);
  k_f32_to_bf16<<<G4, 256, 0, stream>>>(W_hh, Whh_bf, HH, HH);
  k_f32_to_bf16<<<VV, 256, 0, stream>>>(Wv_w, Wv_bf, HH, HH);
  // shifted prev-token embeddings
  k_buildY<<<TT * BB, EE, 0, stream>>>(texts, start_idx, emb, Y_bf);
  // o / h0 / c0
  k_init3<<<3 * BB, HH, 0, stream>>>(tv, Wo_w, Wo_b, Wh_w, Wh_b, Wc_w, Wc_b,
                                     o_buf, h_buf, c_buf);
  // step-invariant gate contribution
  k_constgates<<<BB * 2, 1024, 0, stream>>>(tv, o_buf, W_ih, b_ih, b_hh, cg_buf);
  // h0 -> bf16 double buffer + barrier counter reset (every launch)
  k_prep<<<64, 256, 0, stream>>>(h_buf, h_db, bar_ctr);
  // parallel input-side gate GEMM (A async-staged in LDS)
  k_pregates<<<dim3(4, 256), 256, 0, stream>>>(Y_bf, Wy_bf, pre_g);
  // recurrence: 16 cooperating workgroups, 1 barrier/step
  k_lstm_mw<<<NWG, 128, 0, stream>>>(pre_g, cg_buf, Whh_bf, c_buf, h_db,
                                     bar_ctr, hs_bf);
  // output projection + bias + tbh->btv scatter (A async-staged in LDS)
  k_logits<<<dim3(20, 256), 256, 0, stream>>>(hs_bf, Wv_bf, Wv_b, out);
}
